// SEGRN_40896678593045
// MI455X (gfx1250) — matrix operand order repaired and checked by reading, not compiled, not run on hardware
//
#include <hip/hip_runtime.h>
#include <hip/hip_bf16.h>

// ---------------- problem constants ----------------
#define B_   4
#define C_   256
#define H_   256
#define W_   256
#define H4_  64
#define W4_  64
#define PN_  4096      // 64*64 pooled pixels
#define SG_  8
#define PN4_ 1024      // PN/4
#define NB_  32        // B_*SG_  (all-batch "column" dimension)

typedef __attribute__((ext_vector_type(16))) __bf16 v16bf;
typedef __attribute__((ext_vector_type(8)))  float  v8f;

// native f32 -> bf16 (RNE) — lowers to v_cvt_pk_bf16_f32 pairs on gfx1250
__device__ __forceinline__ __bf16 f2bf(float f) { return (__bf16)f; }

// ---------------- 1) 4x4 mean pool: x[B,C,256,256] -> xp[B,C,64,64] ----------------
// 4 pooled outputs per thread -> one b128 store.
__global__ void pool4x4_kernel(const float* __restrict__ x, float* __restrict__ xp) {
  int i = blockIdx.x * blockDim.x + threadIdx.x;          // over B*C*64*16
  if (i >= B_ * C_ * H4_ * (W4_ / 4)) return;
  int w16 = (i & 15) * 4;                                 // first of 4 w4 outputs
  int h4 = (i >> 4) & 63;
  int bc = i >> 10;                                       // b*C + c
  const float* src = x + ((size_t)(bc * H_) + h4 * 4) * W_ + w16 * 4;
  float ax = 0.f, ay = 0.f, az = 0.f, aw = 0.f;
#pragma unroll
  for (int r = 0; r < 4; ++r) {
    const float4* rowp = (const float4*)(src + (size_t)r * W_);
    float4 a = rowp[0], b = rowp[1], c = rowp[2], d = rowp[3];
    ax += a.x + a.y + a.z + a.w;
    ay += b.x + b.y + b.z + b.w;
    az += c.x + c.y + c.z + c.w;
    aw += d.x + d.y + d.z + d.w;
  }
  float4 res = {ax * (1.f/16.f), ay * (1.f/16.f), az * (1.f/16.f), aw * (1.f/16.f)};
  ((float4*)xp)[i] = res;
}

// ---------------- 2) grouped channel means: xp -> xpre[32][4096] ----------------
__global__ void xpre_kernel(const float* __restrict__ xp, float* __restrict__ xpre) {
  int i = blockIdx.x * blockDim.x + threadIdx.x;          // over 32*4096
  if (i >= NB_ * PN_) return;
  int p = i & (PN_ - 1);
  int n = i >> 12;                                        // b*8+g
  int b = n >> 3, g = n & 7;
  const float* base = xp + ((size_t)(b * C_ + g * 32)) * PN_ + p;
  float acc = 0.f;
#pragma unroll 8
  for (int l = 0; l < 32; ++l) acc += base[(size_t)l * PN_];
  xpre[i] = acc * (1.0f / 32.0f);
}

// ---------------- 3) skinny WMMA GEMM, N=32, transposed store ----------------
// OutT[n][m] = sum_k A[m][k] * Xr[n][k] + bias[m];  A,Xr f32 -> bf16 in regs.
// One wave per 16-row M tile; blockDim = 64 (2 waves).
__global__ __launch_bounds__(64) void gemm_skinny_T(
    const float* __restrict__ A, const float* __restrict__ Xr,
    const float* __restrict__ bias, float* __restrict__ OutT, int M, int K) {
  int wave = blockIdx.x * 2 + (threadIdx.x >> 5);
  int m0 = wave * 16;
  if (m0 >= M) return;                       // wave-uniform: EXEC stays all-ones
  int lane = threadIdx.x & 31;
  int half = lane >> 4;
  int l16  = lane & 15;
  int akb  = half * 8;                       // unused: A also uses kofs
  int kofs = half * 16;                      // B K-sub-base inside 32-chunk
  const float* Arow = A + (size_t)(m0 + l16) * K;
  const float* Bc0  = Xr + (size_t)l16 * K;         // column n = l16
  const float* Bc1  = Xr + (size_t)(16 + l16) * K;  // column n = 16+l16
  v8f acc0 = {};
  v8f acc1 = {};
  for (int kt = 0; kt < K; kt += 32) {
    v16bf a, b0, b1;
#pragma unroll
    for (int j = 0; j < 16; ++j) {
      a[j]     = f2bf(Arow[kt + kofs + j]);
    }
#pragma unroll
    for (int j = 0; j < 16; ++j) {
      b0[j] = f2bf(Bc0[kt + kofs + j]);
      b1[j] = f2bf(Bc1[kt + kofs + j]);
    }
    __builtin_prefetch(Arow + kt + 64, 0, 3);
    acc0 = __builtin_amdgcn_wmma_f32_16x16x32_bf16(false, a, false, b0,
                                                   (short)0, acc0, false, false);
    acc1 = __builtin_amdgcn_wmma_f32_16x16x32_bf16(false, a, false, b1,
                                                   (short)0, acc1, false, false);
  }
  // D layout: lane element r -> (M = m0 + r + 8*half, N = l16 or 16+l16).
  // Per lane the 8 outputs per row are contiguous and 32B-aligned: b128 stores.
  const float* bp = bias + m0 + 8 * half;
  float* o0 = OutT + (size_t)l16 * M + m0 + 8 * half;
  float* o1 = OutT + (size_t)(16 + l16) * M + m0 + 8 * half;
  float4 q0 = {acc0[0] + bp[0], acc0[1] + bp[1], acc0[2] + bp[2], acc0[3] + bp[3]};
  float4 q1 = {acc0[4] + bp[4], acc0[5] + bp[5], acc0[6] + bp[6], acc0[7] + bp[7]};
  float4 p0 = {acc1[0] + bp[0], acc1[1] + bp[1], acc1[2] + bp[2], acc1[3] + bp[3]};
  float4 p1 = {acc1[4] + bp[4], acc1[5] + bp[5], acc1[6] + bp[6], acc1[7] + bp[7]};
  ((float4*)o0)[0] = q0; ((float4*)o0)[1] = q1;
  ((float4*)o1)[0] = p0; ((float4*)o1)[1] = p1;
}

// ---------------- 4) wide WMMA GEMM, 16x64 tile per wave ----------------
// Out[i][j] = sum_k A[i][k]*B'(k,j) (+ biasJ[j])
// B_ROWMAJOR=1: B'(k,j)=Bm[k*N+j]  (support = vT x gcn_weight)
// B_ROWMAJOR=0: B'(k,j)=Bm[j*K+k]  (keyT with kw: columns stored as rows)
template <bool B_ROWMAJOR, bool HAS_BIAS>
__global__ __launch_bounds__(64) void gemm_wide(
    const float* __restrict__ A, const float* __restrict__ Bm,
    const float* __restrict__ biasJ, float* __restrict__ Out,
    int Mrows, int N, int K) {
  int wave = blockIdx.x * 2 + (threadIdx.x >> 5);
  int mtiles = Mrows >> 4;
  int mt = wave % mtiles, nt = wave / mtiles;
  int m0 = mt * 16, n0 = nt * 64;
  if (n0 >= N) return;
  int lane = threadIdx.x & 31;
  int half = lane >> 4;
  int l16  = lane & 15;
  int akb  = half * 8;
  int kofs = half * 16;
  const float* Arow = A + (size_t)(m0 + l16) * K;
  v8f acc[4] = {};
  for (int kt = 0; kt < K; kt += 32) {
    v16bf a;
#pragma unroll
    for (int j = 0; j < 16; ++j) {
      a[j]     = f2bf(Arow[kt + kofs + j]);
    }
    __builtin_prefetch(Arow + kt + 64, 0, 3);
#pragma unroll
    for (int q = 0; q < 4; ++q) {
      v16bf b;
      if (B_ROWMAJOR) {
        const float* bp = Bm + (size_t)(kt + kofs) * N + n0 + q * 16 + l16;
#pragma unroll
        for (int j = 0; j < 16; ++j) b[j] = f2bf(bp[(size_t)j * N]);
      } else {
        const float* bp = Bm + (size_t)(n0 + q * 16 + l16) * K + kt + kofs;
#pragma unroll
        for (int j = 0; j < 16; ++j) b[j] = f2bf(bp[j]);
      }
      acc[q] = __builtin_amdgcn_wmma_f32_16x16x32_bf16(false, a, false, b,
                                                       (short)0, acc[q], false, false);
    }
  }
#pragma unroll
  for (int q = 0; q < 4; ++q) {
    int col = n0 + q * 16 + l16;
    float bj = HAS_BIAS ? biasJ[col] : 0.f;
#pragma unroll
    for (int r = 0; r < 8; ++r) {            // lane-coalesced across the wave
      int i = m0 + r + half * 8;
      Out[(size_t)i * N + col] = acc[q][r] + bj;
    }
  }
}

// ---------------- 5) adjacency: adj[b,s,t] = softmax_t(k[.,s] . q[.,t]) ----------------
__global__ __launch_bounds__(64) void adj_kernel(const float* __restrict__ kT,
                                                 const float* __restrict__ qT,
                                                 float* __restrict__ adj) {
  int b = blockIdx.x;                      // 4 blocks x 64 threads
  int s = threadIdx.x >> 3, t = threadIdx.x & 7;
  const float4* kr = (const float4*)(kT + (size_t)(b * 8 + s) * PN4_);
  const float4* qr = (const float4*)(qT + (size_t)(b * 8 + t) * PN4_);
  float acc = 0.f;
#pragma unroll 4
  for (int d = 0; d < PN4_ / 4; ++d) {
    float4 kv = kr[d], qv = qr[d];
    acc += kv.x * qv.x + kv.y * qv.y + kv.z * qv.z + kv.w * qv.w;
  }
  __shared__ float lg[64];
  lg[threadIdx.x] = acc;
  __syncthreads();
  float mx = -3.4e38f;
  for (int tt = 0; tt < 8; ++tt) mx = fmaxf(mx, lg[s * 8 + tt]);
  float sum = 0.f;
  for (int tt = 0; tt < 8; ++tt) sum += __expf(lg[s * 8 + tt] - mx);
  adj[b * 64 + s * 8 + t] = __expf(acc - mx) / sum;
}

// ---------------- 6) gcn_out[n][o] = sum_t adj[b,s,t]*support[b*8+t][o] + gcn_bias[o] ----------------
__global__ void gcnout_kernel(const float* __restrict__ adj,
                              const float* __restrict__ support,
                              const float* __restrict__ gcn_bias,
                              float* __restrict__ gcn_out) {
  int i = blockIdx.x * blockDim.x + threadIdx.x;          // over 32*4096
  if (i >= NB_ * PN_) return;
  int o = i & (PN_ - 1);
  int n = i >> 12;
  int b = n >> 3, s = n & 7;
  float acc = gcn_bias[o];
  const float* arow = adj + b * 64 + s * 8;
  const float* sb = support + (size_t)b * 8 * PN_ + o;
#pragma unroll
  for (int t = 0; t < 8; ++t) acc += arow[t] * sb[(size_t)t * PN_];
  gcn_out[i] = acc;
}

// ---------------- 7) atten[b,c,s] = softmax_s(keyT[b,c,:] . query[:,s]) ----------------
__global__ __launch_bounds__(256) void atten_kernel(const float* __restrict__ keyT,
                                                    const float* __restrict__ queryT,
                                                    float* __restrict__ atten) {
  int row = blockIdx.x * 32 + (threadIdx.x >> 3);   // b*256 + c
  int s = threadIdx.x & 7;
  int b = row >> 8;
  const float4* kr = (const float4*)(keyT + (size_t)row * PN4_);
  const float4* qr = (const float4*)(queryT + (size_t)(b * 8 + s) * PN4_);
  float acc = 0.f;
#pragma unroll 4
  for (int d = 0; d < PN4_ / 4; ++d) {
    float4 kv = kr[d], qv = qr[d];
    acc += kv.x * qv.x + kv.y * qv.y + kv.z * qv.z + kv.w * qv.w;
  }
  __shared__ float lg[256];
  lg[threadIdx.x] = acc;
  __syncthreads();
  int base = threadIdx.x & ~7;
  float mx = -3.4e38f;
  for (int ss = 0; ss < 8; ++ss) mx = fmaxf(mx, lg[base + ss]);
  float sum = 0.f;
  for (int ss = 0; ss < 8; ++ss) sum += __expf(lg[base + ss] - mx);
  atten[(size_t)row * 8 + s] = __expf(acc - mx) / sum;
}

// ---------------- 8) y[b,c,p] = sum_s atten[b,c,s] * valueT[b*8+s][p] ----------------
// One thread per (b,p); loops over all 256 channels. valueT is read once into
// registers; the per-(b, p-chunk) atten slice (8 KB) is staged into LDS with
// CDNA5 async global->LDS loads and re-read as LDS broadcasts in the c-loop.
__global__ __launch_bounds__(256) void ymix_kernel(const float* __restrict__ atten,
                                                   const float* __restrict__ valueT,
                                                   float* __restrict__ y) {
  int b  = blockIdx.x >> 4;                 // 64 blocks: b in [0,4), pc in [0,16)
  int pc = blockIdx.x & 15;
  int tid = threadIdx.x;
  int p = pc * 256 + tid;

  __shared__ float att_sh[256 * 8];         // atten[b, c=tid, 0..7]
  // async-stage atten slice: thread tid fetches 32B (8 floats) for channel c=tid
  {
    unsigned lds0 = (unsigned)(size_t)(&att_sh[0]) + (unsigned)(tid * 32);
    unsigned long long ga =
        (unsigned long long)(const void*)(atten + ((size_t)b * 256 + tid) * 8);
    asm volatile("global_load_async_to_lds_b128 %0, %1, off"
                 :: "v"(lds0), "v"(ga) : "memory");
    asm volatile("global_load_async_to_lds_b128 %0, %1, off offset:16"
                 :: "v"(lds0), "v"(ga) : "memory");
    asm volatile("s_wait_asynccnt 0" ::: "memory");
  }
  // valueT values for this p, all 8 s: registers for the whole c-loop
  float v[8];
#pragma unroll
  for (int s = 0; s < 8; ++s)
    v[s] = valueT[((size_t)b * 8 + s) * PN_ + p];
  __syncthreads();

  float* yo = y + (size_t)b * 256 * PN_ + p;
  for (int c = 0; c < 256; ++c) {
    const float* at = &att_sh[c * 8];
    float acc = 0.f;
#pragma unroll
    for (int s = 0; s < 8; ++s) acc += at[s] * v[s];
    yo[(size_t)c * PN_] = acc;
  }
}

// ---------------- 9) bilinear upsample 64x64 -> 256x256, align_corners ----------------
// 4 output pixels per thread -> one b128 store (biggest writer: 268 MB).
__global__ void upsample_kernel(const float* __restrict__ y, float* __restrict__ out) {
  int i = blockIdx.x * blockDim.x + threadIdx.x;    // over B*C*256*64
  if (i >= B_ * C_ * H_ * (W_ / 4)) return;
  int ox0 = (i & 63) * 4;
  int oy = (i >> 6) & 255;
  int bc = i >> 14;
  const float SCL = 63.0f / 255.0f;
  float fy = oy * SCL; int y0 = (int)fy; float wy = fy - (float)y0;
  int y1 = (y0 + 1 < 64) ? y0 + 1 : 63;
  const float* r0p = y + (size_t)bc * PN_ + y0 * 64;
  const float* r1p = y + (size_t)bc * PN_ + y1 * 64;
  float4 res;
  float* rp = (float*)&res;
#pragma unroll
  for (int j = 0; j < 4; ++j) {
    int ox = ox0 + j;
    float fx = ox * SCL; int x0 = (int)fx; float wx = fx - (float)x0;
    int x1 = (x0 + 1 < 64) ? x0 + 1 : 63;
    float v00 = r0p[x0], v01 = r0p[x1];
    float v10 = r1p[x0], v11 = r1p[x1];
    float a = v00 + (v10 - v00) * wy;     // interp along y first (matches reference)
    float bb = v01 + (v11 - v01) * wy;
    rp[j] = a + (bb - a) * wx;
  }
  ((float4*)out)[i] = res;
}

// ---------------- launcher ----------------
extern "C" void kernel_launch(void* const* d_in, const int* in_sizes, int n_in,
                              void* d_out, int out_size, void* d_ws, size_t ws_size,
                              hipStream_t stream) {
  const float* x          = (const float*)d_in[0];
  // d_in[1] = se_groups (int scalar) — hardcoded SG_=8
  const float* gcn_weight = (const float*)d_in[2];
  const float* gcn_bias   = (const float*)d_in[3];
  const float* gcn_qw     = (const float*)d_in[4];
  const float* gcn_qb     = (const float*)d_in[5];
  const float* gcn_kw     = (const float*)d_in[6];
  const float* gcn_kb     = (const float*)d_in[7];
  const float* gcn_vw     = (const float*)d_in[8];
  const float* gcn_vb     = (const float*)d_in[9];
  const float* qw         = (const float*)d_in[10];
  const float* qb         = (const float*)d_in[11];
  const float* kw         = (const float*)d_in[12];
  const float* kb         = (const float*)d_in[13];
  const float* vw         = (const float*)d_in[14];
  const float* vb         = (const float*)d_in[15];
  float* out = (float*)d_out;

  // workspace carve-out (floats); total ~10.2M floats ≈ 41 MB
  float* ws = (float*)d_ws;
  size_t o = 0;
  float* xp      = ws + o; o += (size_t)B_ * C_ * PN_;   // 16 MB
  float* xpre    = ws + o; o += (size_t)NB_ * PN_;       // [32][4096]
  float* qT      = ws + o; o += (size_t)NB_ * PN4_;      // [32][1024]
  float* kT      = ws + o; o += (size_t)NB_ * PN4_;
  float* vT      = ws + o; o += (size_t)NB_ * PN_;       // [32][4096]
  float* adj     = ws + o; o += (size_t)B_ * 64;
  float* support = ws + o; o += (size_t)NB_ * PN_;       // [32][4096]
  float* gcnout  = ws + o; o += (size_t)NB_ * PN_;       // [32][4096]
  float* queryT  = ws + o; o += (size_t)NB_ * PN4_;
  float* valueT  = ws + o; o += (size_t)NB_ * PN_;       // [32][4096]
  float* keyT    = ws + o; o += (size_t)PN4_ * PN4_;     // [1024][1024] (rows = b*256+c)
  float* atten   = ws + o; o += (size_t)B_ * C_ * SG_;
  float* ybuf    = ws + o; o += (size_t)B_ * C_ * PN_;   // 16 MB
  (void)ws_size; (void)in_sizes; (void)n_in; (void)out_size;

  // 1-2: pool + grouped means
  pool4x4_kernel<<<(B_ * C_ * H4_ * (W4_ / 4) + 255) / 256, 256, 0, stream>>>(x, xp);
  xpre_kernel<<<(NB_ * PN_ + 255) / 256, 256, 0, stream>>>(xp, xpre);

  // 3: GCN q/k/v skinny GEMMs (bf16 WMMA)
  gemm_skinny_T<<<PN4_ / 32, 64, 0, stream>>>(gcn_qw, xpre, gcn_qb, qT, PN4_, PN_);
  gemm_skinny_T<<<PN4_ / 32, 64, 0, stream>>>(gcn_kw, xpre, gcn_kb, kT, PN4_, PN_);
  gemm_skinny_T<<<PN_  / 32, 64, 0, stream>>>(gcn_vw, xpre, gcn_vb, vT, PN_, PN_);

  // 4: adjacency softmax (8x8 per batch)
  adj_kernel<<<B_, 64, 0, stream>>>(kT, qT, adj);

  // 5: support = vT x gcn_weight  (M=32, N=4096, K=4096; B row-major, no bias)
  gemm_wide<true, false><<<((NB_ / 16) * (PN_ / 64)) / 2, 64, 0, stream>>>(
      vT, gcn_weight, nullptr, support, NB_, PN_, PN_);

  // 6: gcn_out = adj @ support + gcn_bias
  gcnout_kernel<<<(NB_ * PN_ + 255) / 256, 256, 0, stream>>>(adj, support, gcn_bias, gcnout);

  // 7: query / valueT from gcn_out
  gemm_skinny_T<<<PN4_ / 32, 64, 0, stream>>>(qw, gcnout, qb, queryT, PN4_, PN_);
  gemm_skinny_T<<<PN_  / 32, 64, 0, stream>>>(vw, gcnout, vb, valueT, PN_, PN_);

  // 8: keyT = xp x kw^T + kb  (M=1024 rows b*256+c, N=1024, K=4096, biased)
  gemm_wide<false, true><<<((PN4_ / 16) * (PN4_ / 64)) / 2, 64, 0, stream>>>(
      xp, kw, kb, keyT, PN4_, PN4_, PN_);

  // 9-11: attention softmax, value mix (async-LDS staged), bilinear upsample
  atten_kernel<<<B_ * C_ / 32, 256, 0, stream>>>(keyT, queryT, atten);
  ymix_kernel<<<B_ * (PN_ / 256), 256, 0, stream>>>(atten, valueT, ybuf);
  upsample_kernel<<<(B_ * C_ * H_ * (W_ / 4) + 255) / 256, 256, 0, stream>>>(ybuf, out);
}
